// SiteWiseLSTM_26663156973637
// MI455X (gfx1250) — compile-verified, matrix operand order checked by reading, not run on hardware
//
#include <hip/hip_runtime.h>
#include <hip/hip_bf16.h>

typedef __attribute__((ext_vector_type(16))) _Float16 v16h;
typedef __attribute__((ext_vector_type(8)))  _Float16 v8h;
typedef __attribute__((ext_vector_type(8)))  float    v8f;

#define WPB   8      // waves per block (2 waves per SIMD32 at 1 block/WGP)
#define NSEQ  16384  // B*S = 32*512
#define LSEQ  168
#define SDIM  512
#define HID   64

union FragU { v16h v; v8h h[2]; };

__device__ __forceinline__ v8f wmma_acc(v16h a, v16h b, v8f c) {
  return __builtin_amdgcn_wmma_f32_16x16x32_f16(false, a, false, b, (short)0, c,
                                                false, false);
}
__device__ __forceinline__ v8f wmma_z(v16h a, v16h b) {
  v8f z = {};   // SRC2 lowers to inline constant 0
  return __builtin_amdgcn_wmma_f32_16x16x32_f16(false, a, false, b, (short)0, z,
                                                false, false);
}

// Load a 16x32 f16 fragment (A layout / symmetric B layout) from an LDS
// row-major matrix with leading dim `ldk` halves.
//   elems 0..7  -> K = koff + grp*8 + 0..7       (16 contiguous bytes)
//   elems 8..15 -> K = koff + 16 + grp*8 + 0..7  (16 contiguous bytes)
__device__ __forceinline__ v16h ld_frag(const _Float16* base, int row, int ldk,
                                        int koff, int grp) {
  FragU f;
  const _Float16* q = base + row * ldk + koff + grp * 8;
  f.h[0] = *(const v8h*)(q);
  f.h[1] = *(const v8h*)(q + 16);
  return f.v;
}

__device__ __forceinline__ float sigm(float x) {
  return 1.0f / (1.0f + __expf(-x));
}
__device__ __forceinline__ float tanh_fast(float x) {
  return 2.0f / (1.0f + __expf(-2.0f * x)) - 1.0f;
}

// LSTM elementwise with fused bias add: update c tile (C-layout, 8 f32 regs),
// write new h (f16) into the wave-private LDS buffer as row-major 16x64.
__device__ __forceinline__ void ew_store(const v8f& gi, const v8f& gf,
                                         const v8f& gg, const v8f& go,
                                         float bi, float bf, float bg, float bo,
                                         v8f& c, _Float16* hbuf,
                                         int j, int l15, int grp) {
  int col = j * 16 + l15;
#pragma unroll
  for (int v = 0; v < 8; ++v) {
    float cn = sigm(gf[v] + bf) * c[v] + sigm(gi[v] + bi) * tanh_fast(gg[v] + bg);
    c[v] = cn;
    float h = sigm(go[v] + bo) * tanh_fast(cn);
    hbuf[(v + 8 * grp) * HID + col] = (_Float16)h;
  }
}

// Cell with K=64 recurrent GEMM (h @ W^T) + rank-1 scalar-input update.
__device__ __forceinline__ void cell_k64(const _Float16* sW, const float* sBias,
                                         const float* sWin,
                                         v16h hf0, v16h hf1, const float* xb,
                                         v8f* c, _Float16* hbuf,
                                         int l15, int grp) {
#pragma unroll
  for (int j = 0; j < 4; ++j) {
    v8f acc[4];
    float bb[4];
#pragma unroll
    for (int gate = 0; gate < 4; ++gate) {
      int n = gate * 64 + j * 16;
      bb[gate] = sBias[n + l15];
      float w  = sWin[n + l15];
      v8f a = wmma_z(hf0, ld_frag(sW, n + l15, HID, 0, grp));
      a = wmma_acc(hf1, ld_frag(sW, n + l15, HID, 32, grp), a);
#pragma unroll
      for (int v = 0; v < 8; ++v) a[v] += xb[v] * w;   // x (D=1) rank-1 term
      acc[gate] = a;
    }
    ew_store(acc[0], acc[1], acc[2], acc[3], bb[0], bb[1], bb[2], bb[3],
             c[j], hbuf, j, l15, grp);
  }
}

// Layer-1 cell: K=128 GEMM over [h0 | h1] with W = [W_ih1 | W_hh1] (256x128).
__device__ __forceinline__ void cell_k128(const _Float16* sW, const float* sBias,
                                          v16h a0, v16h a1, v16h a2, v16h a3,
                                          v8f* c, _Float16* hbuf,
                                          int l15, int grp) {
#pragma unroll
  for (int j = 0; j < 4; ++j) {
    v8f acc[4];
    float bb[4];
#pragma unroll
    for (int gate = 0; gate < 4; ++gate) {
      int n = gate * 64 + j * 16;
      bb[gate] = sBias[n + l15];
      v8f a = wmma_z(a0, ld_frag(sW, n + l15, 128, 0, grp));
      a = wmma_acc(a1, ld_frag(sW, n + l15, 128, 32, grp), a);
      a = wmma_acc(a2, ld_frag(sW, n + l15, 128, 64, grp), a);
      a = wmma_acc(a3, ld_frag(sW, n + l15, 128, 96, grp), a);
      acc[gate] = a;
    }
    ew_store(acc[0], acc[1], acc[2], acc[3], bb[0], bb[1], bb[2], bb[3],
             c[j], hbuf, j, l15, grp);
  }
}

__global__ __launch_bounds__(WPB * 32) void
sitewise_lstm_fused(const float* __restrict__ x,
                    const float* __restrict__ W_ih0, const float* __restrict__ W_hh0,
                    const float* __restrict__ b_ih0, const float* __restrict__ b_hh0,
                    const float* __restrict__ W_ih1, const float* __restrict__ W_hh1,
                    const float* __restrict__ b_ih1, const float* __restrict__ b_hh1,
                    const float* __restrict__ Wc_ih, const float* __restrict__ Wc_hh,
                    const float* __restrict__ bc_ih, const float* __restrict__ bc_hh,
                    const float* __restrict__ W_out, const float* __restrict__ b_out,
                    const int* __restrict__ horizon_p,
                    float* __restrict__ out) {
  // ---- LDS: f16 weights (shared by all waves) + per-wave h buffer ----
  __shared__ _Float16 sWhh0[256 * 64];    // 32 KB
  __shared__ _Float16 sW1[256 * 128];     // 64 KB  [W_ih1 | W_hh1]
  __shared__ _Float16 sWchh[256 * 64];    // 32 KB
  __shared__ float sB0[256], sB1[256], sBd[256];
  __shared__ float sWih0[256], sWcih[256], sWout[64];
  __shared__ _Float16 sH[WPB][16 * HID];  // 16 KB: one buffer, reused by both layers

  const int tid = threadIdx.x;
  for (int i = tid; i < 256 * 64; i += WPB * 32) sWhh0[i] = (_Float16)W_hh0[i];
  for (int i = tid; i < 256 * 128; i += WPB * 32) {
    int n = i >> 7, k = i & 127;
    sW1[i] = (_Float16)(k < 64 ? W_ih1[n * 64 + k] : W_hh1[n * 64 + k - 64]);
  }
  for (int i = tid; i < 256 * 64; i += WPB * 32) sWchh[i] = (_Float16)Wc_hh[i];
  for (int i = tid; i < 256; i += WPB * 32) {
    sB0[i] = b_ih0[i] + b_hh0[i];
    sB1[i] = b_ih1[i] + b_hh1[i];
    sBd[i] = bc_ih[i] + bc_hh[i];
    sWih0[i] = W_ih0[i];   // (4H,1) column
    sWcih[i] = Wc_ih[i];
  }
  for (int i = tid; i < 64; i += WPB * 32) sWout[i] = W_out[i];
  __syncthreads();

  const int lane = tid & 31;
  const int wave = tid >> 5;
  const int l15 = lane & 15;
  const int grp = lane >> 4;
  const int m0 = (blockIdx.x * WPB + wave) * 16;
  const int row  = m0 + l15;           // batch row in [0, 16384)
  const int bidx = row >> 9;           // b
  const int sidx = row & (SDIM - 1);   // s
  const float* xrow = x + (size_t)bidx * (LSEQ * SDIM) + sidx;

  _Float16* hb = &sH[wave][0];

  v16h h0f0 = {}, h0f1 = {}, h1f0 = {}, h1f1 = {};
  v8f c0[4], c1[4];
#pragma unroll
  for (int j = 0; j < 4; ++j) { c0[j] = (v8f)(0.0f); c1[j] = (v8f)(0.0f); }

  // ---------------- Encoder: 168 fused 2-layer steps ----------------
  float xcur = xrow[0];                 // lane p holds x[row m0 + p%16]
#pragma unroll 1
  for (int t = 0; t < LSEQ; ++t) {
    // Launder a zero offset each iteration so weight B-fragments cannot be
    // hoisted out of the loop (LICM->spill-to-scratch regression in round 2):
    // the loads must stay as per-iteration ds_load_b128 from LDS.
    int lz = 0;
    asm volatile("" : "+v"(lz));
    const _Float16* w0 = sWhh0 + lz;
    const _Float16* w1 = sW1 + lz;

    // software-pipelined prefetch of next timestep's input
    const int tn = (t + 1 < LSEQ) ? t + 1 : (LSEQ - 1);
    float xnext = xrow[tn * SDIM];

    float xb[8];
#pragma unroll
    for (int v = 0; v < 8; ++v) xb[v] = __shfl(xcur, v + 8 * grp, 32);

    cell_k64(w0, sB0, sWih0, h0f0, h0f1, xb, c0, hb, l15, grp);
    h0f0 = ld_frag(hb, l15, HID, 0, grp);
    h0f1 = ld_frag(hb, l15, HID, 32, grp);

    cell_k128(w1, sB1, h0f0, h0f1, h1f0, h1f1, c1, hb, l15, grp);
    h1f0 = ld_frag(hb, l15, HID, 0, grp);
    h1f1 = ld_frag(hb, l15, HID, 32, grp);

    xcur = xnext;
  }

  // ---------------- Decoder: horizon steps ----------------
  const int hor = *horizon_p;
  const float bo = b_out[0];
  float xc = xcur;                      // x_cur = x[:, -1, :]

#pragma unroll 1
  for (int tt = 0; tt < hor; ++tt) {
    int lz = 0;
    asm volatile("" : "+v"(lz));
    const _Float16* wd = sWchh + lz;

    float xb[8];
#pragma unroll
    for (int v = 0; v < 8; ++v) xb[v] = __shfl(xc, v + 8 * grp, 32);

    cell_k64(wd, sBd, sWcih, h1f0, h1f1, xb, c1, hb, l15, grp);
    h1f0 = ld_frag(hb, l15, HID, 0, grp);
    h1f1 = ld_frag(hb, l15, HID, 32, grp);

    // y = h1 . W_out + b_out  (per-lane partial dot over A-frag K chunks)
    float p = 0.0f;
#pragma unroll
    for (int e = 0; e < 8; ++e) p += (float)h1f0[e]     * sWout[grp * 8 + e];
#pragma unroll
    for (int e = 0; e < 8; ++e) p += (float)h1f0[8 + e] * sWout[16 + grp * 8 + e];
#pragma unroll
    for (int e = 0; e < 8; ++e) p += (float)h1f1[e]     * sWout[32 + grp * 8 + e];
#pragma unroll
    for (int e = 0; e < 8; ++e) p += (float)h1f1[8 + e] * sWout[48 + grp * 8 + e];
    p += __shfl_xor(p, 16, 32);
    float y = p + bo;
    xc = y;

    if (lane < 16) {
      out[(size_t)bidx * hor * SDIM + (size_t)tt * SDIM + sidx] = y;
    }
  }
}

extern "C" void kernel_launch(void* const* d_in, const int* in_sizes, int n_in,
                              void* d_out, int out_size, void* d_ws, size_t ws_size,
                              hipStream_t stream) {
  (void)in_sizes; (void)n_in; (void)out_size; (void)d_ws; (void)ws_size;
  const float* x      = (const float*)d_in[0];
  const float* W_ih0  = (const float*)d_in[1];
  const float* W_hh0  = (const float*)d_in[2];
  const float* b_ih0  = (const float*)d_in[3];
  const float* b_hh0  = (const float*)d_in[4];
  const float* W_ih1  = (const float*)d_in[5];
  const float* W_hh1  = (const float*)d_in[6];
  const float* b_ih1  = (const float*)d_in[7];
  const float* b_hh1  = (const float*)d_in[8];
  const float* Wc_ih  = (const float*)d_in[9];
  const float* Wc_hh  = (const float*)d_in[10];
  const float* bc_ih  = (const float*)d_in[11];
  const float* bc_hh  = (const float*)d_in[12];
  const float* W_out  = (const float*)d_in[13];
  const float* b_out  = (const float*)d_in[14];
  const int*   hor    = (const int*)d_in[15];
  float* out = (float*)d_out;

  dim3 grid(NSEQ / (16 * WPB));   // 128 blocks (~1 per WGP)
  dim3 block(WPB * 32);           // 256 threads = 8 waves
  sitewise_lstm_fused<<<grid, block, 0, stream>>>(
      x, W_ih0, W_hh0, b_ih0, b_hh0, W_ih1, W_hh1, b_ih1, b_hh1,
      Wc_ih, Wc_hh, bc_ih, bc_hh, W_out, b_out, hor, out);
}
